// TransformerPredictor_15092515078170
// MI455X (gfx1250) — matrix operand order repaired and checked by reading, not compiled, not run on hardware
//
#include <hip/hip_runtime.h>
#include <hip/hip_bf16.h>

// ---------------------------------------------------------------------------
// CDNA5 (gfx1250) WMMA transformer predictor.
// All heavy math runs through v_wmma_f32_16x16x32_bf16 (bf16 in, f32 acc).
// GEMM staging uses gfx1250 async global->LDS copies (ASYNCcnt) when available.
// ---------------------------------------------------------------------------

typedef __bf16 bf16_t;
typedef bf16_t v16bf __attribute__((ext_vector_type(16)));
typedef float  v8f   __attribute__((ext_vector_type(8)));

union Frag { unsigned u[8]; v16bf v; };

__device__ __forceinline__ unsigned short f32_to_bf16(float f) {
  unsigned u = __builtin_bit_cast(unsigned, f);
  unsigned r = u + 0x7FFFu + ((u >> 16) & 1u);   // round-to-nearest-even
  return (unsigned short)(r >> 16);
}
__device__ __forceinline__ unsigned pack2(unsigned short lo, unsigned short hi) {
  return (unsigned)lo | ((unsigned)hi << 16);
}

#if defined(__gfx1250__) && __has_builtin(__builtin_amdgcn_global_load_async_to_lds_b128) && __has_builtin(__builtin_amdgcn_s_wait_asynccnt)
#define USE_ASYNC_LDS 1
typedef int v4i_vs __attribute__((vector_size(16)));
typedef __attribute__((address_space(1))) v4i_vs as1_v4i;   // global int4
typedef __attribute__((address_space(3))) v4i_vs as3_v4i;   // LDS int4
#else
#define USE_ASYNC_LDS 0
#endif

// ---------------------------------------------------------------------------
// Generic GEMM: C[M,N] = act(A[M,K] * W[K,N] + bias [+ resid]), f32 in/out,
// bf16 WMMA inner product. BM=128, BN=128, BK=32, 8 waves per block.
// Wave (wr = w&3, wc = w>>2) computes a 32x64 output patch as
// 2 A-frags x 4 B-frags = 8 WMMAs per k-step.
// ---------------------------------------------------------------------------
#define AST 34   // padded LDS row stride (u16) to dodge bank conflicts
#define BST 34

__global__ __launch_bounds__(256) void gemm_wmma_kernel(
    const float* __restrict__ A, const float* __restrict__ W,
    const float* __restrict__ bias, const float* __restrict__ resid,
    float* __restrict__ C, int M, int N, int K, int gelu)
{
  __shared__ unsigned short As[128 * AST];   // bf16 A tile [128][32]
  __shared__ unsigned short Bs[128 * BST];   // bf16 W tile transposed [n][k]
#if USE_ASYNC_LDS
  __shared__ float Af[128 * 32];             // f32 staging (async dest)
  __shared__ float Bf[32 * 128];
#endif

  const int tid  = threadIdx.x;
  const int lane = tid & 31;
  const int wave = tid >> 5;
  const int wr   = wave & 3;    // 0..3 -> row offset 32*wr
  const int wc   = wave >> 2;   // 0..1 -> col offset 64*wc
  const int lm   = lane & 15;
  const int half = (lane < 16) ? 0 : 8;
  const int m0 = blockIdx.y * 128;
  const int n0 = blockIdx.x * 128;

  v8f acc0[4] = {};
  v8f acc1[4] = {};

  for (int k0 = 0; k0 < K; k0 += 32) {
    if (k0 + 32 < K) {  // prefetch next k tile -> global_prefetch_b8
      __builtin_prefetch(A + (size_t)(m0 + (tid >> 1)) * K + (k0 + 32) + (tid & 1) * 16);
      __builtin_prefetch(W + (size_t)(k0 + 32 + (tid >> 3)) * N + n0 + (tid & 7) * 16);
    }
#if USE_ASYNC_LDS
    // Async copy f32 tiles into LDS staging (b128 = 4 floats per lane per op)
#pragma unroll
    for (int i = 0; i < 4; ++i) {
      int idx4 = tid + i * 256;              // A tile: 1024 float4 chunks
      int r = idx4 >> 3, c4 = idx4 & 7;
      const float* g = A + (size_t)(m0 + r) * K + k0 + c4 * 4;
      __builtin_amdgcn_global_load_async_to_lds_b128(
          (as1_v4i*)g, (as3_v4i*)(Af + idx4 * 4), 0, 0);
    }
#pragma unroll
    for (int i = 0; i < 4; ++i) {
      int idx4 = tid + i * 256;              // W tile: 1024 float4 chunks
      int kk = idx4 >> 5, n4 = idx4 & 31;
      const float* g = W + (size_t)(k0 + kk) * N + n0 + n4 * 4;
      __builtin_amdgcn_global_load_async_to_lds_b128(
          (as1_v4i*)g, (as3_v4i*)(Bf + idx4 * 4), 0, 0);
    }
    __builtin_amdgcn_s_wait_asynccnt(0);
    __syncthreads();
    // cooperative f32 -> bf16 conversion into padded WMMA tiles
    for (int i = tid; i < 128 * 32; i += 256) {
      As[(i >> 5) * AST + (i & 31)] = f32_to_bf16(Af[i]);
      Bs[(i & 127) * BST + (i >> 7)] = f32_to_bf16(Bf[i]);
    }
#else
    // Direct staging: load f32, convert, store bf16
    for (int i = tid; i < 128 * 32; i += 256) {
      int r = i >> 5, c = i & 31;
      As[r * AST + c] = f32_to_bf16(A[(size_t)(m0 + r) * K + (k0 + c)]);
    }
    for (int i = tid; i < 32 * 128; i += 256) {
      int kk = i >> 7, nn = i & 127;
      Bs[nn * BST + kk] = f32_to_bf16(W[(size_t)(k0 + kk) * N + (n0 + nn)]);
    }
#endif
    __syncthreads();

    // A fragments (16x32 each): lane m = lm, kbase 0/8 per half, VGPR4-7 = K+16
    Frag fa0, fa1;
    {
      const int kb2 = ((lane < 16) ? 0 : 8) >> 1;
      const unsigned* p0 = (const unsigned*)(As + (wr * 32 + lm) * AST);
      const unsigned* p1 = (const unsigned*)(As + (wr * 32 + 16 + lm) * AST);
#pragma unroll
      for (int i = 0; i < 4; ++i) {
        fa0.u[i] = p0[kb2 + i];  fa0.u[4 + i] = p0[kb2 + 8 + i];
        fa1.u[i] = p1[kb2 + i];  fa1.u[4 + i] = p1[kb2 + 8 + i];
      }
    }
    // B fragments (32x16 each): lane n = lm, kbase 0/8 per half, VGPR4-7 = K+16
    Frag fb[4];
    {
      const int kb2 = ((lane < 16) ? 0 : 8) >> 1;
#pragma unroll
      for (int t = 0; t < 4; ++t) {
        const unsigned* p = (const unsigned*)(Bs + (wc * 64 + 16 * t + lm) * BST);
#pragma unroll
        for (int i = 0; i < 8; ++i) fb[t].u[i] = p[kb2 + i + (i & 4)];
      }
    }
#pragma unroll
    for (int t = 0; t < 4; ++t) {
      acc0[t] = __builtin_amdgcn_wmma_f32_16x16x32_bf16(false, fa0.v, false, fb[t].v, (short)0, acc0[t], false, false);
      acc1[t] = __builtin_amdgcn_wmma_f32_16x16x32_bf16(false, fa1.v, false, fb[t].v, (short)0, acc1[t], false, false);
    }
    __syncthreads();
  }

  // Epilogue: C/D layout -> lane n = lm, row = half + vgpr index
  const int row0 = m0 + wr * 32 + half;
#pragma unroll
  for (int t = 0; t < 4; ++t) {
    const int col = n0 + wc * 64 + 16 * t + lm;
#pragma unroll
    for (int r = 0; r < 8; ++r) {
      {
        int row = row0 + r;
        float v = acc0[t][r] + bias[col];
        if (resid) v += resid[(size_t)row * N + col];
        if (gelu)  v = 0.5f * v * (1.0f + erff(v * 0.70710678118654752f));
        C[(size_t)row * N + col] = v;
      }
      {
        int row = row0 + 16 + r;
        float v = acc1[t][r] + bias[col];
        if (resid) v += resid[(size_t)row * N + col];
        if (gelu)  v = 0.5f * v * (1.0f + erff(v * 0.70710678118654752f));
        C[(size_t)row * N + col] = v;
      }
    }
  }
}

// ---------------------------------------------------------------------------
// LayerNorm over D=384. One block (128 threads) per output row.
// compact=1 remaps output row r -> input row (r/256)*512 + 256 + (r%256).
// ---------------------------------------------------------------------------
__global__ __launch_bounds__(128) void ln_kernel(
    const float* __restrict__ in, float* __restrict__ out,
    const float* __restrict__ w, const float* __restrict__ b, int compact)
{
  __shared__ float red[128];
  const int row  = blockIdx.x;
  const int irow = compact ? ((row >> 8) * 512 + 256 + (row & 255)) : row;
  const float* x = in + (size_t)irow * 384;
  const int t = threadIdx.x;

  float v0 = x[t], v1 = x[t + 128], v2 = x[t + 256];
  red[t] = v0 + v1 + v2;
  __syncthreads();
  for (int o = 64; o > 0; o >>= 1) { if (t < o) red[t] += red[t + o]; __syncthreads(); }
  float mu = red[0] * (1.0f / 384.0f);
  __syncthreads();
  float d0 = v0 - mu, d1 = v1 - mu, d2 = v2 - mu;
  red[t] = d0 * d0 + d1 * d1 + d2 * d2;
  __syncthreads();
  for (int o = 64; o > 0; o >>= 1) { if (t < o) red[t] += red[t + o]; __syncthreads(); }
  float rstd = rsqrtf(red[0] * (1.0f / 384.0f) + 1e-5f);

  float* y = out + (size_t)row * 384;
  y[t]       = d0 * rstd * w[t]       + b[t];
  y[t + 128] = d1 * rstd * w[t + 128] + b[t + 128];
  y[t + 256] = d2 * rstd * w[t + 256] + b[t + 256];
}

// ---------------------------------------------------------------------------
// Build h[b, s, d]: s<256 -> ctx tile xe[masks_x[b][s%8]];
//                  s>=256 -> mask_token + pos[masks[b][(s-256)%32]]
// ---------------------------------------------------------------------------
__global__ __launch_bounds__(256) void build_h_kernel(
    const float* __restrict__ xe, const float* __restrict__ pos,
    const int* __restrict__ masks_x, const int* __restrict__ masks,
    const float* __restrict__ mask_token, float* __restrict__ h)
{
  int idx = blockIdx.x * 256 + threadIdx.x;
  if (idx >= 16 * 512 * 384) return;
  int d = idx % 384;
  int s = (idx / 384) & 511;
  int b = idx / (384 * 512);
  float v;
  if (s < 256) {
    int node = masks_x[b * 8 + (s & 7)];
    v = xe[(size_t)node * 384 + d];
  } else {
    int tt = masks[b * 32 + ((s - 256) & 31)];
    v = mask_token[d] + pos[(size_t)tt * 384 + d];
  }
  h[idx] = v;
}

// ---------------------------------------------------------------------------
// Flash attention per (graph b, head hh). S=512, HD=64, scale folded into Q.
// K rows + V^T staged as bf16 in dynamic LDS; 8 waves x 16 queries per pass,
// 4 passes. Online softmax with __shfl_xor reductions over the 16-lane halves
// (matches the WMMA C-matrix layout). P@V via WMMA.
// ---------------------------------------------------------------------------
#define KST 66    // K rows padded: [512][66] u16
#define VST 514   // V^T rows padded: [64][514] u16
#define PST 34    // per-wave P tile rows: [16][34] u16

__global__ __launch_bounds__(256) void attn_kernel(
    const float* __restrict__ qkv, float* __restrict__ out)
{
  extern __shared__ unsigned short smem[];
  unsigned short* Ks = smem;                 // 512*KST
  unsigned short* Vt = smem + 512 * KST;     // 64*VST
  unsigned short* Ps = Vt + 64 * VST;        // 8*16*PST

  const int b  = blockIdx.x / 6;
  const int hh = blockIdx.x % 6;
  const int tid  = threadIdx.x;
  const int lane = tid & 31;
  const int wave = tid >> 5;
  const int lm   = lane & 15;
  const int half = (lane < 16) ? 0 : 8;
  const size_t base = (size_t)b * 512 * 1152;

  // Stage K (row-major) and V (transposed) as bf16
  for (int i = tid; i < 512 * 64; i += 256) {
    int s = i >> 6, d = i & 63;
    const float* rowp = qkv + base + (size_t)s * 1152 + hh * 64;
    Ks[s * KST + d]  = f32_to_bf16(rowp[384 + d]);
    Vt[d * VST + s]  = f32_to_bf16(rowp[768 + d]);
  }
  __syncthreads();

  unsigned short* Pw = Ps + wave * 16 * PST;

  for (int qi = 0; qi < 4; ++qi) {
    const int qbase = qi * 128 + wave * 16;

    // Q fragments (scale folded in): A layout, kd 0..31 and 32..63
    Frag aQ0, aQ1;
    {
      const float* qp = qkv + base + (size_t)(qbase + lm) * 1152 + hh * 64;
      const int kb = (lane < 16) ? 0 : 8;
#pragma unroll
      for (int i = 0; i < 8; ++i) {
        int ke = kb + ((i < 4) ? 2 * i : 16 + 2 * (i - 4));
        aQ0.u[i] = pack2(f32_to_bf16(qp[ke] * 0.125f),      f32_to_bf16(qp[ke + 1] * 0.125f));
        aQ1.u[i] = pack2(f32_to_bf16(qp[32 + ke] * 0.125f), f32_to_bf16(qp[32 + ke + 1] * 0.125f));
      }
    }

    float mrun[8], ssum[8];
#pragma unroll
    for (int r = 0; r < 8; ++r) { mrun[r] = -1e30f; ssum[r] = 0.0f; }
    v8f O0 = {}, O1 = {}, O2 = {}, O3 = {};

    for (int j = 0; j < 512; j += 32) {
      const int kb2h = ((lane < 16) ? 0 : 8) >> 1;
      Frag bk00, bk01, bk10, bk11;
      {
        const unsigned* p0 = (const unsigned*)(Ks + (j + lm) * KST);
        const unsigned* p1 = (const unsigned*)(Ks + (j + 16 + lm) * KST);
#pragma unroll
        for (int i = 0; i < 8; ++i) {
          bk00.u[i] = p0[kb2h + i + (i & 4)];       bk01.u[i] = p0[16 + kb2h + i + (i & 4)];
          bk10.u[i] = p1[kb2h + i + (i & 4)];       bk11.u[i] = p1[16 + kb2h + i + (i & 4)];
        }
      }
      v8f s0 = {}, s1 = {};
      s0 = __builtin_amdgcn_wmma_f32_16x16x32_bf16(false, aQ0.v, false, bk00.v, (short)0, s0, false, false);
      s0 = __builtin_amdgcn_wmma_f32_16x16x32_bf16(false, aQ1.v, false, bk01.v, (short)0, s0, false, false);
      s1 = __builtin_amdgcn_wmma_f32_16x16x32_bf16(false, aQ0.v, false, bk10.v, (short)0, s1, false, false);
      s1 = __builtin_amdgcn_wmma_f32_16x16x32_bf16(false, aQ1.v, false, bk11.v, (short)0, s1, false, false);

      // online softmax update + stash P tile in LDS
#pragma unroll
      for (int r = 0; r < 8; ++r) {
        float t = fmaxf(s0[r], s1[r]);
        t = fmaxf(t, __shfl_xor(t, 1, 32));
        t = fmaxf(t, __shfl_xor(t, 2, 32));
        t = fmaxf(t, __shfl_xor(t, 4, 32));
        t = fmaxf(t, __shfl_xor(t, 8, 32));
        float mnew = fmaxf(mrun[r], t);
        float f  = __expf(mrun[r] - mnew);
        float p0 = __expf(s0[r] - mnew);
        float p1 = __expf(s1[r] - mnew);
        ssum[r] = ssum[r] * f + p0 + p1;
        O0[r] *= f; O1[r] *= f; O2[r] *= f; O3[r] *= f;
        mrun[r] = mnew;
        int prow = half + r;
        Pw[prow * PST + lm]      = f32_to_bf16(p0);
        Pw[prow * PST + 16 + lm] = f32_to_bf16(p1);
      }

      // P fragment (A layout)
      Frag aP;
      {
        const int kb2 = ((lane < 16) ? 0 : 8) >> 1;
        const unsigned* p = (const unsigned*)(Pw + lm * PST);
#pragma unroll
        for (int i = 0; i < 4; ++i) aP.u[i]     = p[kb2 + i];
#pragma unroll
        for (int i = 0; i < 4; ++i) aP.u[4 + i] = p[kb2 + 8 + i];
      }

      // O += P @ V[j:j+32, :], 4 column tiles of 16
      {
        const unsigned* p0 = (const unsigned*)(Vt + (0 * 16 + lm) * VST + j);
        const unsigned* p1 = (const unsigned*)(Vt + (1 * 16 + lm) * VST + j);
        const unsigned* p2 = (const unsigned*)(Vt + (2 * 16 + lm) * VST + j);
        const unsigned* p3 = (const unsigned*)(Vt + (3 * 16 + lm) * VST + j);
        Frag bv0, bv1, bv2, bv3;
#pragma unroll
        for (int i = 0; i < 8; ++i) {
          bv0.u[i] = p0[kb2h + i + (i & 4)]; bv1.u[i] = p1[kb2h + i + (i & 4)];
          bv2.u[i] = p2[kb2h + i + (i & 4)]; bv3.u[i] = p3[kb2h + i + (i & 4)];
        }
        O0 = __builtin_amdgcn_wmma_f32_16x16x32_bf16(false, aP.v, false, bv0.v, (short)0, O0, false, false);
        O1 = __builtin_amdgcn_wmma_f32_16x16x32_bf16(false, aP.v, false, bv1.v, (short)0, O1, false, false);
        O2 = __builtin_amdgcn_wmma_f32_16x16x32_bf16(false, aP.v, false, bv2.v, (short)0, O2, false, false);
        O3 = __builtin_amdgcn_wmma_f32_16x16x32_bf16(false, aP.v, false, bv3.v, (short)0, O3, false, false);
      }
    }

    // normalize and write out
#pragma unroll
    for (int r = 0; r < 8; ++r) {
      float t = ssum[r];
      t += __shfl_xor(t, 1, 32);
      t += __shfl_xor(t, 2, 32);
      t += __shfl_xor(t, 4, 32);
      t += __shfl_xor(t, 8, 32);
      float inv = 1.0f / t;
      float* op = out + ((size_t)(b * 512 + qbase + half + r)) * 384 + hh * 64;
      op[lm]      = O0[r] * inv;
      op[16 + lm] = O1[r] * inv;
      op[32 + lm] = O2[r] * inv;
      op[48 + lm] = O3[r] * inv;
    }
  }
}

// ---------------------------------------------------------------------------
// Host-side orchestration
// ---------------------------------------------------------------------------
static inline void launch_gemm(const float* A, const float* W, const float* bias,
                               const float* resid, float* C,
                               int M, int N, int K, int gelu, hipStream_t s)
{
  dim3 grid(N / 128, M / 128);
  gemm_wmma_kernel<<<grid, 256, 0, s>>>(A, W, bias, resid, C, M, N, K, gelu);
}

extern "C" void kernel_launch(void* const* d_in, const int* in_sizes, int n_in,
                              void* d_out, int out_size, void* d_ws, size_t ws_size,
                              hipStream_t stream) {
  const float* x          = (const float*)d_in[0];
  const float* target_pe  = (const float*)d_in[1];
  const int*   masks_x    = (const int*)  d_in[2];
  const int*   masks      = (const int*)  d_in[3];
  // d_in[4] = batch_idx (unused in node-node path)
  const float* W_embed    = (const float*)d_in[5];
  const float* b_embed    = (const float*)d_in[6];
  const float* W_pos      = (const float*)d_in[7];
  const float* b_pos      = (const float*)d_in[8];
  const float* mask_token = (const float*)d_in[9];
  const float* ln1_w      = (const float*)d_in[10];
  const float* ln1_b      = (const float*)d_in[11];
  const float* qkv_w      = (const float*)d_in[12];
  const float* qkv_b      = (const float*)d_in[13];
  const float* proj_w     = (const float*)d_in[14];
  const float* proj_b     = (const float*)d_in[15];
  const float* ln2_w      = (const float*)d_in[16];
  const float* ln2_b      = (const float*)d_in[17];
  const float* fc1_w      = (const float*)d_in[18];
  const float* fc1_b      = (const float*)d_in[19];
  const float* fc2_w      = (const float*)d_in[20];
  const float* fc2_b      = (const float*)d_in[21];
  const float* lnf_w      = (const float*)d_in[22];
  const float* lnf_b      = (const float*)d_in[23];
  const float* W_out      = (const float*)d_in[24];
  const float* b_out      = (const float*)d_in[25];

  float* ws = (float*)d_ws;
  size_t o = 0;
  float* h    = ws + o; o += (size_t)8192 * 384;
  float* a    = ws + o; o += (size_t)8192 * 384;
  float* qkv  = ws + o; o += (size_t)8192 * 1152;
  float* attn = ws + o; o += (size_t)8192 * 384;
  float* mlp  = ws + o; o += (size_t)8192 * 1536;
  float* xe   = ws + o; o += (size_t)4096 * 384;
  float* pos  = ws + o; o += (size_t)2048 * 384;

  // 1) predictor_embed: xe = x @ W_embed + b_embed    [4096,512]x[512,384]
  launch_gemm(x, W_embed, b_embed, nullptr, xe, 4096, 384, 512, 0, stream);
  // 2) predictor_pos_embed: pos = target_pe @ W_pos + b_pos   [2048,64]x[64,384]
  launch_gemm(target_pe, W_pos, b_pos, nullptr, pos, 2048, 384, 64, 0, stream);
  // 3) build token sequence h [16,512,384]
  build_h_kernel<<<(16 * 512 * 384 + 255) / 256, 256, 0, stream>>>(
      xe, pos, masks_x, masks, mask_token, h);

  const size_t attn_lds = (size_t)(512 * KST + 64 * VST + 8 * 16 * PST) * sizeof(unsigned short);

  for (int l = 0; l < 3; ++l) {
    const float* qw = qkv_w  + (size_t)l * 384 * 1152;
    const float* qb = qkv_b  + (size_t)l * 1152;
    const float* pw = proj_w + (size_t)l * 384 * 384;
    const float* pb = proj_b + (size_t)l * 384;
    const float* f1w = fc1_w + (size_t)l * 384 * 1536;
    const float* f1b = fc1_b + (size_t)l * 1536;
    const float* f2w = fc2_w + (size_t)l * 1536 * 384;
    const float* f2b = fc2_b + (size_t)l * 384;

    ln_kernel<<<8192, 128, 0, stream>>>(h, a, ln1_w + l * 384, ln1_b + l * 384, 0);
    launch_gemm(a, qw, qb, nullptr, qkv, 8192, 1152, 384, 0, stream);
    attn_kernel<<<16 * 6, 256, attn_lds, stream>>>(qkv, attn);
    launch_gemm(attn, pw, pb, h, h, 8192, 384, 384, 0, stream);      // h += attn @ proj
    ln_kernel<<<8192, 128, 0, stream>>>(h, a, ln2_w + l * 384, ln2_b + l * 384, 0);
    launch_gemm(a, f1w, f1b, nullptr, mlp, 8192, 1536, 384, 1, stream);   // GELU
    launch_gemm(mlp, f2w, f2b, h, h, 8192, 384, 1536, 0, stream);    // h += mlp @ fc2
  }

  // final LN restricted to pred tokens (compacted to [4096,384])
  ln_kernel<<<4096, 128, 0, stream>>>(h, a, lnf_w, lnf_b, 1);
  // output projection: [4096,384] x [384,512] -> d_out
  launch_gemm(a, W_out, b_out, nullptr, (float*)d_out, 4096, 512, 384, 0, stream);
}
